// TwoDimEquivalent_51926154609382
// MI455X (gfx1250) — compile-verified, matrix operand order checked by reading, not run on hardware
//
#include <hip/hip_runtime.h>

typedef __attribute__((ext_vector_type(16))) _Float16     v16h;
typedef __attribute__((ext_vector_type(8)))  float        v8f;
typedef __attribute__((ext_vector_type(4)))  unsigned int u32x4;
typedef __attribute__((ext_vector_type(8)))  int          i32x8;
typedef __attribute__((ext_vector_type(4)))  int          i32x4;

#ifndef M_PI
#define M_PI 3.14159265358979323846
#endif

// CDNA5 has a hardware TRANS tanh (V_TANH_F32): 1 TRANS/node instead of the
// 2 (exp+rcp) needed by the sech^2-via-exp formulation.
#if __has_builtin(__builtin_amdgcn_tanhf)
#define DEV_TANHF(x) __builtin_amdgcn_tanhf(x)
#define HAVE_HW_TANH 1
#elif __has_builtin(__builtin_amdgcn_tanh_f32)
#define DEV_TANHF(x) __builtin_amdgcn_tanh_f32(x)
#define HAVE_HW_TANH 1
#else
#define HAVE_HW_TANH 0
#endif

#if __has_builtin(__builtin_amdgcn_exp2f)
#define DEV_EXP2F(x) __builtin_amdgcn_exp2f(x)
#else
#define DEV_EXP2F(x) exp2f(x)
#endif

#if __has_builtin(__builtin_amdgcn_rcpf)
#define DEV_RCPF(x) __builtin_amdgcn_rcpf(x)
#else
#define DEV_RCPF(x) (1.0f / (x))
#endif

#if __has_builtin(__builtin_amdgcn_tensor_load_to_lds)
#define HAVE_TDM 1
#if __has_include(<hip/amd_detail/amd_gfx1250_TDM.h>)
#define TDM_SIX_ARGS 1   // therock-10.0 headers => 6-arg builtin arity
#endif
#else
#define HAVE_TDM 0
#endif

#define CS 32            // timesteps per chunk
#define ROWS 16          // batch rows per wave
#define LPAD 33          // LDS row stride in dwords (32 + 1 pad -> conflict-free)

// ---------------------------------------------------------------------------
// Init kernel: 32 positive roots of P_64 (Gauss-Legendre NQ=64) + folded
// effective weights, double precision Newton.  xw[0..31]=A*x_k,
// xw[32..63]=2*w_k*A*exp(-(A x_k)^2/2)/sqrt(2pi)  (factor 2 folds -x_k).
// ---------------------------------------------------------------------------
__global__ __launch_bounds__(32) void gl64_init_kernel(float* __restrict__ xw) {
    const int i = threadIdx.x;
    const double n = 64.0;
    double x = cos(M_PI * (i + 0.75) / (n + 0.5));
    double p0, p1, dp = 1.0;
    #pragma unroll 1
    for (int it = 0; it < 30; ++it) {
        p0 = 1.0; p1 = x;
        #pragma unroll 1
        for (int k = 2; k <= 64; ++k) {
            double pk = ((2.0 * k - 1.0) * x * p1 - (k - 1.0) * p0) / (double)k;
            p0 = p1; p1 = pk;
        }
        dp = n * (x * p1 - p0) / (x * x - 1.0);
        x  = x - p1 / dp;
    }
    p0 = 1.0; p1 = x;
    #pragma unroll 1
    for (int k = 2; k <= 64; ++k) {
        double pk = ((2.0 * k - 1.0) * x * p1 - (k - 1.0) * p0) / (double)k;
        p0 = p1; p1 = pk;
    }
    dp = n * (x * p1 - p0) / (x * x - 1.0);
    const double w    = 2.0 / ((1.0 - x * x) * dp * dp);
    const double A    = 5.0;
    const double xs   = A * x;
    const double weff = 2.0 * w * A * exp(-xs * xs * 0.5) / sqrt(2.0 * M_PI);
    xw[i]      = (float)xs;
    xw[32 + i] = (float)weff;
}

// ---------------------------------------------------------------------------
// TDM: async-load a [16 rows x 32 cols] f32 tile of u (row stride = T elems)
// into LDS with 1-dword row padding (LDS row stride = 33 dwords).
// D# built per CDNA5 ISA §8.3/8.4; 2D tensor => groups 2/3 zero.
// ---------------------------------------------------------------------------
__device__ __forceinline__ void tdm_load_u_tile(unsigned long long gaddr,
                                                unsigned lds_off, int T) {
#if HAVE_TDM
    u32x4 g0;
    g0.x = 1u;                                             // count=1, user D#
    g0.y = lds_off;                                        // lds_addr
    g0.z = (unsigned)(gaddr & 0xFFFFFFFFull);              // global_addr[31:0]
    g0.w = (unsigned)((gaddr >> 32) & 0x1FFFFFFull)        // global_addr[56:32]
         | (2u << 30);                                     // type=2 (image)
    i32x8 g1;
    g1[0] = (int)((2u << 16)                               // data_size=4B
                | (1u << 20)                               // pad_enable
                | (4u << 22)                               // pad_interval: 32 dwords
                | (0u << 25));                             // pad_amount: 1 dword
    g1[1] = (int)((unsigned)CS << 16);                     // tensor_dim0[15:0]=32
    g1[2] = (int)((unsigned)ROWS << 16);                   // tensor_dim1[15:0]=16
    g1[3] = (int)((unsigned)CS << 16);                     // tile_dim0=32
    g1[4] = ROWS;                                          // tile_dim1=16, tile_dim2=0
    g1[5] = T;                                             // tensor_dim0_stride[31:0]
    g1[6] = 0;
    g1[7] = 0;
    i32x4 z4 = {0, 0, 0, 0};
#ifdef TDM_SIX_ARGS
    i32x8 z8 = {0, 0, 0, 0, 0, 0, 0, 0};
    __builtin_amdgcn_tensor_load_to_lds(g0, g1, z4, z4, z8, 0);
#else
    __builtin_amdgcn_tensor_load_to_lds(g0, g1, z4, z4, 0);
#endif
#else
    (void)gaddr; (void)lds_off; (void)T;
#endif
}

__device__ __forceinline__ void tdm_wait(int pending) {
#if HAVE_TDM
    if (pending) __builtin_amdgcn_s_wait_tensorcnt(1);
    else         __builtin_amdgcn_s_wait_tensorcnt(0);
#else
    (void)pending;
#endif
    __asm__ volatile("" ::: "memory");
}

// ---------------------------------------------------------------------------
// One recurrence step: 16 sech^2 TRANS chains per lane feed the B matrix of
// a single v_wmma_f32_16x16x32_f16 (A = ones) that reduces the 32-node
// weighted quadrature per batch row with f32 accumulation.
// ---------------------------------------------------------------------------
__device__ __forceinline__ float step_update(float ut, float& k1, float& k2,
                                             float& v, const float* xk,
                                             const float* wk, v16h a_ones) {
    const float alpha = 0.2f;                       // DT/TAU
    const float delta = sqrtf(__builtin_fmaf(k1, k1,
                              __builtin_fmaf(k2, k2, ut * ut)));
    v16h bmat;
    #pragma unroll
    for (int j = 0; j < 16; ++j) {
        const float y = delta * xk[j];              // >= 0 (positive nodes)
#if HAVE_HW_TANH
        const float th   = DEV_TANHF(y);            // v_tanh_f32: 1 TRANS/node
        const float dphi = __builtin_fmaf(-th, th, 1.0f);
#else
        const float NEG2_LOG2E = -2.885390081777927f; // -2/ln(2)
        const float e    = DEV_EXP2F(NEG2_LOG2E * y);
        const float ri   = DEV_RCPF(1.0f + e);
        const float dphi = 4.0f * e * ri * ri;      // 1 - tanh^2
#endif
        bmat[j] = (_Float16)(wk[j] * dphi);
    }
    v8f c = {};
    c = __builtin_amdgcn_wmma_f32_16x16x32_f16(false, a_ones, false, bmat,
                                               (short)0, c, false, false);
    const float g = c[0];
    k1 = __builtin_fmaf(alpha, __builtin_fmaf(g, __builtin_fmaf(0.5f, v, k1), -k1), k1);
    k2 = __builtin_fmaf(alpha, __builtin_fmaf(g, __builtin_fmaf(0.5f, k2, 1.9f * v), -k2), k2);
    v  = __builtin_fmaf(alpha, ut - v, v);
    return __builtin_fmaf(2.8f, k1, -2.2f * k2);    // z
}

// ---------------------------------------------------------------------------
// Main kernel: one wave <-> 16 batch rows.  Double-buffered TDM tiles of u in
// LDS; z staged in a padded LDS tile and flushed as coalesced 128B rows.
// ---------------------------------------------------------------------------
__global__ __launch_bounds__(32) void rate_recurrence_wmma_kernel(
        const float* __restrict__ u,   // [B, T]
        float* __restrict__ out,       // [B, T+1]
        const float* __restrict__ xw,  // [64]
        int T) {
    __shared__ float ubuf[2][ROWS * LPAD];
    __shared__ float ztile[ROWS * LPAD];

    const int lane = threadIdx.x;
    const int row  = lane & 15;
    const int half = lane >> 4;
    const long long bbase = (long long)blockIdx.x * ROWS;
    const long long b     = bbase + row;

    float xk[16], wk[16];
    #pragma unroll
    for (int j = 0; j < 16; ++j) {
        xk[j] = xw[half * 16 + j];
        wk[j] = xw[32 + half * 16 + j];
    }
    v16h a_ones;
    #pragma unroll
    for (int j = 0; j < 16; ++j) a_ones[j] = (_Float16)1.0f;

    float k1 = 0.0f, k2 = 0.0f, v = 0.0f;
    const float* urow = u + (long long)b * T;
    float*       orow = out + (long long)b * (T + 1);
    if (half == 0) orow[0] = 0.0f;

    const unsigned long long ugbase =
        (unsigned long long)(uintptr_t)(u + (long long)bbase * T);
    const int NC = T / CS;

#if HAVE_TDM
    tdm_load_u_tile(ugbase, (unsigned)(uintptr_t)&ubuf[0][0], T);
#endif

    #pragma unroll 1
    for (int cc = 0; cc < NC; ++cc) {
        const int sel = cc & 1;
#if HAVE_TDM
        if (cc + 1 < NC) {
            tdm_load_u_tile(ugbase + (unsigned long long)(cc + 1) * (CS * 4),
                            (unsigned)(uintptr_t)&ubuf[sel ^ 1][0], T);
            tdm_wait(1);           // oldest (current buffer) done; next in flight
        } else {
            tdm_wait(0);
        }
#else
        // Fallback: cooperative copy of the tile into LDS (single wave; the
        // compiler tracks these plain load/store dependencies itself).
        #pragma unroll
        for (int q = lane; q < ROWS * CS; q += 32) {
            const int r = q >> 5, j = q & 31;
            ubuf[sel][r * LPAD + j] = u[(long long)(bbase + r) * T + cc * CS + j];
        }
        __asm__ volatile("" ::: "memory");
#endif
        volatile const float* up = &ubuf[sel][row * LPAD];
        #pragma unroll 1
        for (int j = 0; j < CS; ++j) {
            const float ut = up[j];                         // broadcast pair read
            const float z  = step_update(ut, k1, k2, v, xk, wk, a_ones);
            ztile[row * LPAD + j] = z;                      // half-pairs: same value
        }
        // Coalesced flush: 16 rows x 32 lanes, one 128B line per store.
        const int tb = cc * CS + 1;
        #pragma unroll
        for (int r = 0; r < ROWS; ++r) {
            out[(long long)(bbase + r) * (T + 1) + tb + lane] = ztile[r * LPAD + lane];
        }
    }

    // Generic tail (T not a multiple of CS): direct path.
    #pragma unroll 1
    for (int t = NC * CS; t < T; ++t) {
        const float ut = urow[t];
        const float z  = step_update(ut, k1, k2, v, xk, wk, a_ones);
        if (half == 0) orow[t + 1] = z;
    }
}

// ---------------------------------------------------------------------------
extern "C" void kernel_launch(void* const* d_in, const int* in_sizes, int n_in,
                              void* d_out, int out_size, void* d_ws, size_t ws_size,
                              hipStream_t stream) {
    (void)n_in; (void)ws_size;
    const float* u   = (const float*)d_in[0];
    float*       out = (float*)d_out;
    float*       xw  = (float*)d_ws;

    const int B = out_size - in_sizes[0];   // B*(T+1) - B*T
    const int T = in_sizes[0] / B;

    gl64_init_kernel<<<1, 32, 0, stream>>>(xw);
    rate_recurrence_wmma_kernel<<<B / ROWS, 32, 0, stream>>>(u, out, xw, T);
}
